// SCARF_89000312307787
// MI455X (gfx1250) — compile-verified, matrix operand order checked by reading, not run on hardware
//
#include <hip/hip_runtime.h>
#include <stdint.h>

// ---------------- WMMA types (CDNA5 gfx1250, wave32) ----------------
typedef __attribute__((ext_vector_type(16))) __bf16 v16bf;
typedef __attribute__((ext_vector_type(8)))  float  v8f;
typedef __attribute__((ext_vector_type(4)))  int    v4i_t;

#define B_ROWS 8192
#define M_IN   512
#define H_DIM  1024

#define BM 128     // block tile rows (batch)
#define BN 256     // block tile cols (features)
#define BK 32      // K step = one WMMA
#define LDSS 40    // LDS row pitch in bf16 elems (32 data + 8 pad -> conflict-free b128 reads)

union Frag {
    uint4 q[2];
    v16bf v;
};

// ---- optional CDNA5 async global->LDS staging (ASYNCcnt-tracked) ----
#if defined(__has_builtin)
#  if __has_builtin(__builtin_amdgcn_global_load_async_to_lds_b128)
#    define HAS_ASYNC_LDS 1
#  endif
#endif

__device__ __forceinline__ void copy16_g2lds(const void* g, void* l) {
#ifdef HAS_ASYNC_LDS
    // Signature (from compiler diagnostic): param0 = v4i* (global/device AS),
    // then LDS dst, imm offset, imm cpol.
    __builtin_amdgcn_global_load_async_to_lds_b128((v4i_t*)g, (v4i_t*)l, 0, 0);
#else
    *(uint4*)l = *(const uint4*)g;
#endif
}

__device__ __forceinline__ void wait_stage() {
#ifdef HAS_ASYNC_LDS
#  if __has_builtin(__builtin_amdgcn_s_wait_asynccnt)
    __builtin_amdgcn_s_wait_asynccnt(0);
#  else
    asm volatile("s_wait_asynccnt 0" ::: "memory");
#  endif
#endif
}

__device__ __forceinline__ unsigned short f32_to_bf16_bits(float f) {
    union { float f; uint32_t u; } cv; cv.f = f;
    uint32_t u = cv.u;
    uint32_t r = u + 0x7FFFu + ((u >> 16) & 1u);   // round-to-nearest-even
    return (unsigned short)(r >> 16);
}

// ---------------- prep: bf16 anchor + corrupted positive ----------------
__global__ void prep_inputs(const float* __restrict__ anchor,
                            const float* __restrict__ rnd,
                            const unsigned char* __restrict__ mask,
                            unsigned short* __restrict__ a16,
                            unsigned short* __restrict__ p16,
                            int n) {
    int i = blockIdx.x * blockDim.x + threadIdx.x;
    if (i >= n) return;
    float a = anchor[i];
    float p = mask[i] ? rnd[i] : a;
    a16[i] = f32_to_bf16_bits(a);
    p16[i] = f32_to_bf16_bits(p);
}

// ---------------- transpose f32 (K x N) -> bf16 (N x K) ----------------
__global__ void transpose_w(const float* __restrict__ W,
                            unsigned short* __restrict__ WT,
                            int K, int N) {
    __shared__ float tile[32][33];
    int k0 = blockIdx.y * 32;
    int n0 = blockIdx.x * 32;
    int tx = threadIdx.x;          // 0..31
    int ty = threadIdx.y;          // 0..7
    #pragma unroll
    for (int r = ty; r < 32; r += 8)
        tile[r][tx] = W[(size_t)(k0 + r) * N + n0 + tx];
    __syncthreads();
    #pragma unroll
    for (int r = ty; r < 32; r += 8)
        WT[(size_t)(n0 + r) * K + k0 + tx] = f32_to_bf16_bits(tile[tx][r]);
}

// ---------------- fused GEMM + bias + ReLU ----------------
// C[Mrows x N] = relu(A[Mrows x K](bf16) @ W[K x N] + bias), W given transposed (N x K, bf16).
// 256 threads = 8 waves (2 x 4); block tile 128x256; wave tile 64x64; double-buffered LDS.
template <bool OUT_F32>
__global__ __launch_bounds__(256)
void gemm_bias_relu(const unsigned short* __restrict__ A,
                    const unsigned short* __restrict__ WT,
                    const float* __restrict__ bias,
                    void* __restrict__ out,
                    int N, int K) {
    __shared__ __align__(16) unsigned short sA[2][BM * LDSS];  // 20 KB
    __shared__ __align__(16) unsigned short sB[2][BN * LDSS];  // 40 KB

    const int tid   = threadIdx.x;
    const int lane  = tid & 31;
    const int wave  = tid >> 5;          // 0..7
    const int waveM = wave >> 2;         // 0..1  (row group of 64)
    const int waveN = wave & 3;          // 0..3  (col group of 64)
    const int lidx  = lane & 15;
    const int lhalf = lane >> 4;

    const int tileN0 = blockIdx.x * BN;
    const int tileM0 = blockIdx.y * BM;

    v8f acc[4][4];
    #pragma unroll
    for (int i = 0; i < 4; ++i)
        #pragma unroll
        for (int j = 0; j < 4; ++j)
            acc[i][j] = v8f{};

    // Stage one 128x32 A tile (512 chunks) + 256x32 W^T tile (1024 chunks), 16B each.
    auto stage = [&](int buf, int k0) {
        #pragma unroll
        for (int it = 0; it < 2; ++it) {
            int c = tid + it * 256;
            int row = c >> 2, seg = c & 3;
            copy16_g2lds(A + (size_t)(tileM0 + row) * K + k0 + seg * 8,
                         &sA[buf][row * LDSS + seg * 8]);
        }
        #pragma unroll
        for (int it = 0; it < 4; ++it) {
            int c = tid + it * 256;
            int row = c >> 2, seg = c & 3;
            copy16_g2lds(WT + (size_t)(tileN0 + row) * K + k0 + seg * 8,
                         &sB[buf][row * LDSS + seg * 8]);
        }
    };

    const int nk = K / BK;
    stage(0, 0);

    for (int ks = 0; ks < nk; ++ks) {
        const int cur = ks & 1;
        wait_stage();                 // async stores into 'cur' complete (this wave)
        __syncthreads();              // all waves staged 'cur'; all done reading 'cur^1'
        if (ks + 1 < nk) stage(cur ^ 1, (ks + 1) * BK);

        // Per-lane 16-bit operand fragments: lane%16 = row (A) / col (B),
        // lane/16 selects K halves: q[0]=K[kh*8..+7], q[1]=K[kh*8+16..+23].
        Frag aF[4];
        #pragma unroll
        for (int i = 0; i < 4; ++i) {
            int r = waveM * 64 + i * 16 + lidx;
            aF[i].q[0] = *(const uint4*)&sA[cur][r * LDSS + lhalf * 8];
            aF[i].q[1] = *(const uint4*)&sA[cur][r * LDSS + lhalf * 8 + 16];
        }
        #pragma unroll
        for (int j = 0; j < 4; ++j) {
            Frag bF;
            int r = waveN * 64 + j * 16 + lidx;
            bF.q[0] = *(const uint4*)&sB[cur][r * LDSS + lhalf * 8];
            bF.q[1] = *(const uint4*)&sB[cur][r * LDSS + lhalf * 8 + 16];
            #pragma unroll
            for (int i = 0; i < 4; ++i)
                acc[i][j] = __builtin_amdgcn_wmma_f32_16x16x32_bf16(
                    false, aF[i].v, false, bF.v,
                    (short)0, acc[i][j], false, false);
        }
    }

    // Epilogue: bias + ReLU. C/D layout: col = lane%16, row = vgpr + 8*(lane/16).
    #pragma unroll
    for (int j = 0; j < 4; ++j) {
        int col  = tileN0 + waveN * 64 + j * 16 + lidx;
        float bj = bias[col];
        #pragma unroll
        for (int i = 0; i < 4; ++i) {
            int row0 = tileM0 + waveM * 64 + i * 16 + lhalf * 8;
            #pragma unroll
            for (int v = 0; v < 8; ++v) {
                float val = acc[i][j][v] + bj;
                val = val > 0.f ? val : 0.f;
                size_t off = (size_t)(row0 + v) * N + col;
                if (OUT_F32) ((float*)out)[off] = val;
                else         ((unsigned short*)out)[off] = f32_to_bf16_bits(val);
            }
        }
    }
}

// ---------------- host-side orchestration ----------------
extern "C" void kernel_launch(void* const* d_in, const int* in_sizes, int n_in,
                              void* d_out, int out_size, void* d_ws, size_t ws_size,
                              hipStream_t stream) {
    const float*         anchor = (const float*)d_in[0];
    const float*         rnd    = (const float*)d_in[1];
    const unsigned char* mask   = (const unsigned char*)d_in[2];  // jnp.bool_ -> 1 byte
    const float*         enc_w0 = (const float*)d_in[3];
    const float*         enc_b0 = (const float*)d_in[4];
    const float*         enc_w  = (const float*)d_in[5];          // (3,H,H)
    const float*         enc_b  = (const float*)d_in[6];          // (3,H)
    const float*         head_w = (const float*)d_in[7];          // (2,H,H)
    const float*         head_b = (const float*)d_in[8];          // (2,H)

    char* ws = (char*)d_ws;
    size_t off = 0;
    auto take = [&](size_t bytes) -> char* {
        char* p = ws + off;
        off += (bytes + 255) & ~(size_t)255;
        return p;
    };
    unsigned short* a16  = (unsigned short*)take((size_t)B_ROWS * M_IN * 2);
    unsigned short* p16  = (unsigned short*)take((size_t)B_ROWS * M_IN * 2);
    unsigned short* actA = (unsigned short*)take((size_t)B_ROWS * H_DIM * 2);
    unsigned short* actB = (unsigned short*)take((size_t)B_ROWS * H_DIM * 2);
    unsigned short* wt0  = (unsigned short*)take((size_t)H_DIM * M_IN * 2);
    unsigned short* wtE[3];
    for (int i = 0; i < 3; ++i) wtE[i] = (unsigned short*)take((size_t)H_DIM * H_DIM * 2);
    unsigned short* wtH[2];
    for (int i = 0; i < 2; ++i) wtH[i] = (unsigned short*)take((size_t)H_DIM * H_DIM * 2);

    // 1) bf16 inputs + corruption
    {
        int n = B_ROWS * M_IN;
        prep_inputs<<<(n + 255) / 256, 256, 0, stream>>>(anchor, rnd, mask, a16, p16, n);
    }
    // 2) transpose+convert all weights
    {
        dim3 blk(32, 8);
        transpose_w<<<dim3(H_DIM / 32, M_IN / 32), blk, 0, stream>>>(enc_w0, wt0, M_IN, H_DIM);
        for (int i = 0; i < 3; ++i)
            transpose_w<<<dim3(H_DIM / 32, H_DIM / 32), blk, 0, stream>>>(
                enc_w + (size_t)i * H_DIM * H_DIM, wtE[i], H_DIM, H_DIM);
        for (int i = 0; i < 2; ++i)
            transpose_w<<<dim3(H_DIM / 32, H_DIM / 32), blk, 0, stream>>>(
                head_w + (size_t)i * H_DIM * H_DIM, wtH[i], H_DIM, H_DIM);
    }

    // 3) two towers of 6 fused GEMM+bias+ReLU layers
    dim3 gg(H_DIM / BN, B_ROWS / BM);   // (4, 64)
    float* outF = (float*)d_out;
    const unsigned short* inputs[2] = { a16, p16 };
    for (int t = 0; t < 2; ++t) {
        const unsigned short* x = inputs[t];
        float* outT = outF + (size_t)t * B_ROWS * H_DIM;
        gemm_bias_relu<false><<<gg, 256, 0, stream>>>(x,    wt0,    enc_b0,             actA, H_DIM, M_IN);
        gemm_bias_relu<false><<<gg, 256, 0, stream>>>(actA, wtE[0], enc_b + 0 * H_DIM,  actB, H_DIM, H_DIM);
        gemm_bias_relu<false><<<gg, 256, 0, stream>>>(actB, wtE[1], enc_b + 1 * H_DIM,  actA, H_DIM, H_DIM);
        gemm_bias_relu<false><<<gg, 256, 0, stream>>>(actA, wtE[2], enc_b + 2 * H_DIM,  actB, H_DIM, H_DIM);
        gemm_bias_relu<false><<<gg, 256, 0, stream>>>(actB, wtH[0], head_b + 0 * H_DIM, actA, H_DIM, H_DIM);
        gemm_bias_relu<true ><<<gg, 256, 0, stream>>>(actA, wtH[1], head_b + 1 * H_DIM, outT, H_DIM, H_DIM);
    }
    (void)in_sizes; (void)n_in; (void)out_size; (void)ws_size;
}
